// CausalSelfAttention_27736898798011
// MI455X (gfx1250) — matrix operand order repaired and checked by reading, not compiled, not run on hardware
//
#include <hip/hip_runtime.h>

// CDNA5 / gfx1250 causal self-attention: f16 WMMA (f32 accumulate) end to end.
// Round 5: double-buffered K fragments across key blocks in attention
// (ping-pong, same pattern as the GEMMs), packed 16B V-transposed stores.

typedef __attribute__((ext_vector_type(16))) _Float16 v16h;
typedef __attribute__((ext_vector_type(8)))  _Float16 v8h;
typedef __attribute__((ext_vector_type(8)))  float    v8f;

#define WMMA_F32_F16(a, b, c) \
  __builtin_amdgcn_wmma_f32_16x16x32_f16(false, (a), false, (b), (short)0, (c), false, false)

static constexpr int Bb = 8, Tt = 2048, Cc = 512, Hh = 8, Dd = 64;
static constexpr int Mrows = Bb * Tt;  // 16384

// ---- fragment loaders (layouts per CDNA5 ISA §7.12.2) ----

// A matrix 16x32 f16, row-major source with leading dim `ld`.
__device__ __forceinline__ v16h load_a16x32(const _Float16* p, int ld, int lane) {
  int m = lane & 15, g = lane >> 4;
  const _Float16* r = p + m * ld + g * 8;
  v8h lo = *(const v8h*)(r);
  v8h hi = *(const v8h*)(r + 16);
  v16h a;
#pragma unroll
  for (int i = 0; i < 8; i++) { a[i] = lo[i]; a[i + 8] = hi[i]; }
  return a;
}

// B matrix 32x16 f16 from Bt stored row-major as [16 (n), K]:  B[k][n] = Bt[n][k].
__device__ __forceinline__ v16h load_bt16(const _Float16* p, int ld, int lane) {
  int n = lane & 15, g = lane >> 4;
  const _Float16* r = p + n * ld + g * 8;
  v8h lo = *(const v8h*)(r);
  v8h hi = *(const v8h*)(r + 16);
  v16h b;
#pragma unroll
  for (int i = 0; i < 8; i++) { b[i] = lo[i]; b[i + 8] = hi[i]; }
  return b;
}

// ---- kernels ----

__global__ void cvt_f32_f16(const float* __restrict__ in, _Float16* __restrict__ out, int n) {
  int i = blockIdx.x * blockDim.x + threadIdx.x;
  int stride = gridDim.x * blockDim.x;
  for (; i < n; i += stride) out[i] = (_Float16)in[i];
}

// Fused QKV projection: y = x @ W.T ; Q,K scattered to [B,H,T,D], V to [B,H,D,T].
// Each wave: 32x64 output strip, ping-pong double-buffered fragments.
// grid = (Mrows/32) * 3 mats * 2 halves ; block = 128 (4 waves, 64 cols each)
__global__ void __launch_bounds__(128, 1)
qkv_gemm(const _Float16* __restrict__ xh,
         const _Float16* __restrict__ Wqh,
         const _Float16* __restrict__ Wkh,
         const _Float16* __restrict__ Wvh,
         _Float16* __restrict__ Qh,
         _Float16* __restrict__ Kh,
         _Float16* __restrict__ Vt) {
  int lane = threadIdx.x & 31, wave = threadIdx.x >> 5;
  int bid = blockIdx.x;
  int mt = bid / 6;
  int r = bid % 6;
  int mat = r >> 1, half = r & 1;
  int n0 = half * 256 + wave * 64;
  const _Float16* W = (mat == 0) ? Wqh : ((mat == 1) ? Wkh : Wvh);
  _Float16* Out = (mat == 0) ? Qh : ((mat == 1) ? Kh : Vt);

  const _Float16* xp = xh + (size_t)mt * 32 * Cc;
  const _Float16* Wb = W + (size_t)n0 * Cc;

  v8f acc[2][4] = {};
  v16h a0[2], b0[4], a1[2], b1[4];

  // prologue: set0 <- k = 0
  a0[0] = load_a16x32(xp, Cc, lane);
  a0[1] = load_a16x32(xp + 16 * Cc, Cc, lane);
#pragma unroll
  for (int t = 0; t < 4; t++) b0[t] = load_bt16(Wb + (size_t)t * 16 * Cc, Cc, lane);

  for (int k = 0; k < Cc; k += 64) {  // 8 iterations, 2 K-steps each
    // set1 <- k+32 (always in range: k+32 <= 480)
    a1[0] = load_a16x32(xp + k + 32, Cc, lane);
    a1[1] = load_a16x32(xp + 16 * Cc + k + 32, Cc, lane);
#pragma unroll
    for (int t = 0; t < 4; t++)
      b1[t] = load_bt16(Wb + (size_t)t * 16 * Cc + k + 32, Cc, lane);
    // consume set0
#pragma unroll
    for (int t = 0; t < 4; t++) {
      acc[0][t] = WMMA_F32_F16(a0[0], b0[t], acc[0][t]);
      acc[1][t] = WMMA_F32_F16(a0[1], b0[t], acc[1][t]);
    }
    // set0 <- k+64 (guarded)
    if (k + 64 < Cc) {
      a0[0] = load_a16x32(xp + k + 64, Cc, lane);
      a0[1] = load_a16x32(xp + 16 * Cc + k + 64, Cc, lane);
#pragma unroll
      for (int t = 0; t < 4; t++)
        b0[t] = load_bt16(Wb + (size_t)t * 16 * Cc + k + 64, Cc, lane);
    }
    // consume set1
#pragma unroll
    for (int t = 0; t < 4; t++) {
      acc[0][t] = WMMA_F32_F16(a1[0], b1[t], acc[0][t]);
      acc[1][t] = WMMA_F32_F16(a1[1], b1[t], acc[1][t]);
    }
  }

  int g = lane >> 4, nlo = lane & 15;
#pragma unroll
  for (int mi = 0; mi < 2; mi++) {
#pragma unroll
    for (int t = 0; t < 4; t++) {
      int n = n0 + t * 16 + nlo;
      int h = n >> 6, d = n & 63;
      int m0 = mt * 32 + mi * 16 + g * 8;  // first of 8 consecutive rows
      int bidx = m0 >> 11, tpos0 = m0 & (Tt - 1);
      if (mat != 2) {  // uniform branch: Q,K -> [B,H,T,D], scalar stores (d-contig across lanes)
#pragma unroll
        for (int j = 0; j < 8; j++)
          Qh[0];  // placeholder removed below
      }
      if (mat != 2) {
#pragma unroll
        for (int j = 0; j < 8; j++)
          Out[((size_t)(bidx * Hh + h) * Tt + tpos0 + j) * Dd + d] = (_Float16)acc[mi][t][j];
      } else {
        // V transposed [B,H,D,T]: 8 consecutive tpos per lane -> one 16B store
        v8h pk;
#pragma unroll
        for (int j = 0; j < 8; j++) pk[j] = (_Float16)acc[mi][t][j];
        *(v8h*)(Out + ((size_t)(bidx * Hh + h) * Dd + d) * Tt + tpos0) = pk;
      }
    }
  }
}

// Flash attention: one wave per 16-query tile, streaming 32-key blocks.
// K fragments double-buffered across key blocks; V from Vt [B,H,D,T].
// grid = B*H * (T/16/4) = 2048 ; block = 128 (4 waves, distinct q tiles)
__global__ void __launch_bounds__(128, 1)
attn(const _Float16* __restrict__ Qh,
     const _Float16* __restrict__ Kh,
     const _Float16* __restrict__ Vt,
     _Float16* __restrict__ Yh) {
  __shared__ __align__(32) _Float16 sP[4][16 * 32];  // P tile per wave

  int lane = threadIdx.x & 31, wave = threadIdx.x >> 5;
  int bh = blockIdx.x >> 5;       // 0..63
  int qg = blockIdx.x & 31;
  int qt = qg * 4 + wave;         // 0..127
  int q0 = qt * 16;

  const _Float16* Qp  = Qh + ((size_t)bh * Tt + q0) * Dd;
  const _Float16* Kp  = Kh + (size_t)bh * Tt * Dd;
  const _Float16* Vtp = Vt + (size_t)bh * Dd * Tt;

  v16h a0 = load_a16x32(Qp, Dd, lane);        // d = 0..31
  v16h a1 = load_a16x32(Qp + 32, Dd, lane);   // d = 32..63

  v8f o[4] = {};
  float mrow[8], lrow[8];
#pragma unroll
  for (int j = 0; j < 8; j++) { mrow[j] = -1e30f; lrow[j] = 0.0f; }

  int g = lane >> 4, nlo = lane & 15;
  const float scale = 0.125f;  // 1/sqrt(64)
  _Float16* pw = sP[wave];
  const int kbEnd = q0 + 15;    // last valid key block start <= kbEnd

  auto loadK = [&](int kb, v16h (&kf)[4]) {
    kf[0] = load_bt16(Kp + (size_t)kb * Dd, Dd, lane);
    kf[1] = load_bt16(Kp + (size_t)kb * Dd + 32, Dd, lane);
    kf[2] = load_bt16(Kp + (size_t)(kb + 16) * Dd, Dd, lane);
    kf[3] = load_bt16(Kp + (size_t)(kb + 16) * Dd + 32, Dd, lane);
  };

  // one key block: scores from preloaded kf, prefetch K for kb+32 into kfn
  auto step = [&](int kb, v16h (&kf)[4], v16h (&kfn)[4]) {
    // S = Q . K^T (fragments already resident)
    v8f s0 = {}, s1 = {};
    s0 = WMMA_F32_F16(a0, kf[0], s0);
    s0 = WMMA_F32_F16(a1, kf[1], s0);
    s1 = WMMA_F32_F16(a0, kf[2], s1);
    s1 = WMMA_F32_F16(a1, kf[3], s1);

    // V fragments for this block; overlap softmax below
    v16h vb[4];
#pragma unroll
    for (int t = 0; t < 4; t++)
      vb[t] = load_bt16(Vtp + (size_t)(t * 16) * Tt + kb, Tt, lane);

    // prefetch next block's K fragments: complete under softmax + PV
    if (kb + 32 <= kbEnd) loadK(kb + 32, kfn);
    if (kb + 64 <= kbEnd)
      __builtin_prefetch((const void*)(Kp + (size_t)(kb + 64 + lane) * Dd), 0, 0);

    bool need_mask = (kb + 31 > q0);
#pragma unroll
    for (int j = 0; j < 8; j++) {
      int qrow = q0 + g * 8 + j;
      float v0 = s0[j] * scale, v1 = s1[j] * scale;
      if (need_mask) {
        if (kb + nlo > qrow) v0 = -1e30f;
        if (kb + 16 + nlo > qrow) v1 = -1e30f;
      }
      float mx = fmaxf(v0, v1);
#pragma unroll
      for (int off = 1; off < 16; off <<= 1) mx = fmaxf(mx, __shfl_xor(mx, off, 32));
      float mnew = fmaxf(mrow[j], mx);
      float alpha = __expf(mrow[j] - mnew);
      mrow[j] = mnew;
      float p0 = __expf(v0 - mnew), p1 = __expf(v1 - mnew);
      float sum = p0 + p1;
#pragma unroll
      for (int off = 1; off < 16; off <<= 1) sum += __shfl_xor(sum, off, 32);
      lrow[j] = lrow[j] * alpha + sum;
      pw[(g * 8 + j) * 32 + nlo] = (_Float16)p0;
      pw[(g * 8 + j) * 32 + 16 + nlo] = (_Float16)p1;
#pragma unroll
      for (int t = 0; t < 4; t++) o[t][j] *= alpha;
    }

    __builtin_amdgcn_wave_barrier();      // LDS P stores visible before re-fragmenting

    v16h pa = load_a16x32(pw, 32, lane);  // P as A fragment 16x32
#pragma unroll
    for (int t = 0; t < 4; t++)
      o[t] = WMMA_F32_F16(pa, vb[t], o[t]);

    __builtin_amdgcn_wave_barrier();  // protect LDS WAR for next iteration
  };

  v16h kf0[4], kf1[4];
  loadK(0, kf0);
  for (int kb = 0; kb <= kbEnd; kb += 64) {
    step(kb, kf0, kf1);          // consumes kf0, preloads kf1 (kb+32)
    if (kb + 32 <= kbEnd)
      step(kb + 32, kf1, kf0);   // consumes kf1, preloads kf0 (kb+64)
  }

  // normalize and write back into [B,T,C] f16 (heads recombined)
  int b = bh >> 3, h = bh & 7;
#pragma unroll
  for (int t = 0; t < 4; t++) {
#pragma unroll
    for (int j = 0; j < 8; j++) {
      int row = q0 + g * 8 + j;
      float val = o[t][j] / lrow[j];
      Yh[(size_t)(b * Tt + row) * Cc + h * Dd + t * 16 + nlo] = (_Float16)val;
    }
  }
}

// Output projection: out = y @ Wp.T, fp32 result. Same ping-pong pipeline.
// grid = (Mrows/32) * 2 ; block = 128
__global__ void __launch_bounds__(128, 1)
proj_gemm(const _Float16* __restrict__ Yh,
          const _Float16* __restrict__ Wph,
          float* __restrict__ out) {
  int lane = threadIdx.x & 31, wave = threadIdx.x >> 5;
  int mt = blockIdx.x >> 1, half = blockIdx.x & 1;
  int n0 = half * 256 + wave * 64;

  const _Float16* yp = Yh + (size_t)mt * 32 * Cc;
  const _Float16* Wb = Wph + (size_t)n0 * Cc;

  v8f acc[2][4] = {};
  v16h a0[2], b0[4], a1[2], b1[4];

  a0[0] = load_a16x32(yp, Cc, lane);
  a0[1] = load_a16x32(yp + 16 * Cc, Cc, lane);
#pragma unroll
  for (int t = 0; t < 4; t++) b0[t] = load_bt16(Wb + (size_t)t * 16 * Cc, Cc, lane);

  for (int k = 0; k < Cc; k += 64) {
    a1[0] = load_a16x32(yp + k + 32, Cc, lane);
    a1[1] = load_a16x32(yp + 16 * Cc + k + 32, Cc, lane);
#pragma unroll
    for (int t = 0; t < 4; t++)
      b1[t] = load_bt16(Wb + (size_t)t * 16 * Cc + k + 32, Cc, lane);
#pragma unroll
    for (int t = 0; t < 4; t++) {
      acc[0][t] = WMMA_F32_F16(a0[0], b0[t], acc[0][t]);
      acc[1][t] = WMMA_F32_F16(a0[1], b0[t], acc[1][t]);
    }
    if (k + 64 < Cc) {
      a0[0] = load_a16x32(yp + k + 64, Cc, lane);
      a0[1] = load_a16x32(yp + 16 * Cc + k + 64, Cc, lane);
#pragma unroll
      for (int t = 0; t < 4; t++)
        b0[t] = load_bt16(Wb + (size_t)t * 16 * Cc + k + 64, Cc, lane);
    }
#pragma unroll
    for (int t = 0; t < 4; t++) {
      acc[0][t] = WMMA_F32_F16(a1[0], b1[t], acc[0][t]);
      acc[1][t] = WMMA_F32_F16(a1[1], b1[t], acc[1][t]);
    }
  }

  int g = lane >> 4, nlo = lane & 15;
#pragma unroll
  for (int mi = 0; mi < 2; mi++)
#pragma unroll
    for (int t = 0; t < 4; t++)
#pragma unroll
      for (int j = 0; j < 8; j++)
        out[(size_t)(mt * 32 + mi * 16 + g * 8 + j) * Cc + n0 + t * 16 + nlo] =
            acc[mi][t][j];
}

extern "C" void kernel_launch(void* const* d_in, const int* in_sizes, int n_in,
                              void* d_out, int out_size, void* d_ws, size_t ws_size,
                              hipStream_t stream) {
  (void)in_sizes; (void)n_in; (void)out_size; (void)ws_size;
  const float* x  = (const float*)d_in[0];
  const float* Wq = (const float*)d_in[1];
  const float* Wk = (const float*)d_in[2];
  const float* Wv = (const float*)d_in[3];
  const float* Wp = (const float*)d_in[4];

  char* ws = (char*)d_ws;
  size_t off = 0;
  auto carve = [&](size_t bytes) -> _Float16* {
    _Float16* p = (_Float16*)(ws + off);
    off += (bytes + 255) & ~(size_t)255;
    return p;
  };
  _Float16* xh  = carve((size_t)Mrows * Cc * 2);   // 16 MB
  _Float16* Wqh = carve((size_t)Cc * Cc * 2);
  _Float16* Wkh = carve((size_t)Cc * Cc * 2);
  _Float16* Wvh = carve((size_t)Cc * Cc * 2);
  _Float16* Wph = carve((size_t)Cc * Cc * 2);
  _Float16* Qh  = carve((size_t)Bb * Hh * Tt * Dd * 2);  // 16 MB each
  _Float16* Kh  = carve((size_t)Bb * Hh * Tt * Dd * 2);
  _Float16* Vt  = carve((size_t)Bb * Hh * Dd * Tt * 2);
  _Float16* Yh  = carve((size_t)Mrows * Cc * 2);

  cvt_f32_f16<<<2048, 256, 0, stream>>>(x, xh, Mrows * Cc);
  cvt_f32_f16<<<256, 256, 0, stream>>>(Wq, Wqh, Cc * Cc);
  cvt_f32_f16<<<256, 256, 0, stream>>>(Wk, Wkh, Cc * Cc);
  cvt_f32_f16<<<256, 256, 0, stream>>>(Wv, Wvh, Cc * Cc);
  cvt_f32_f16<<<256, 256, 0, stream>>>(Wp, Wph, Cc * Cc);

  qkv_gemm<<<(Mrows / 32) * 6, 128, 0, stream>>>(xh, Wqh, Wkh, Wvh, Qh, Kh, Vt);
  attn<<<(Bb * Hh) * (Tt / 16 / 4), 128, 0, stream>>>(Qh, Kh, Vt, Yh);
  proj_gemm<<<(Mrows / 32) * 2, 128, 0, stream>>>(Yh, Wph, (float*)d_out);
}
